// LinearAttention_4054449127645
// MI455X (gfx1250) — matrix operand order repaired and checked by reading, not compiled, not run on hardware
//
#include <hip/hip_runtime.h>

// ---------------------------------------------------------------------------
// MI455X (gfx1250, wave32) implementation of the LinearAttention block.
// Heavy GEMMs (qkv projection, ctx = k^T v) use v_wmma_f32_16x16x32_bf16 with
// double-buffered LDS fed by CDNA5 global_load_async_to_lds_b128 (ASYNCcnt).
// ---------------------------------------------------------------------------

typedef __attribute__((ext_vector_type(4)))  unsigned int u32x4;
typedef __attribute__((ext_vector_type(16))) __bf16       v16bf;
typedef __attribute__((ext_vector_type(8)))  float        v8f;

#define B_     64
#define C_     256
#define N_     256     // H*W
#define HEADS_ 8
#define HD_    256
#define HID_   2048
#define TEMB_  512
#define EPS_   1e-6f
#define SC_    0.0625f // HEAD_DIM^-0.5

// round-to-nearest-even f32 -> bf16 (bit pattern in ushort)
__device__ __forceinline__ unsigned short f2bf(float f) {
  unsigned int u = __float_as_uint(f);
  u += 0x7FFFu + ((u >> 16) & 1u);
  return (unsigned short)(u >> 16);
}

__device__ __forceinline__ float wred_max(float x) {
  #pragma unroll
  for (int o = 16; o > 0; o >>= 1) x = fmaxf(x, __shfl_xor(x, o, 32));
  return x;
}
__device__ __forceinline__ float wred_sum(float x) {
  #pragma unroll
  for (int o = 16; o > 0; o >>= 1) x += __shfl_xor(x, o, 32);
  return x;
}

// CDNA5 async global->LDS copy, 16B per lane. vdst VGPR holds the
// wave-relative LDS byte address (= low 32 bits of the flat LDS pointer).
__device__ __forceinline__ void async_ld_b128(unsigned lds_addr, const void* gaddr) {
  asm volatile("global_load_async_to_lds_b128 %0, %1, off"
               :: "v"(lds_addr), "v"(gaddr) : "memory");
}
__device__ __forceinline__ void wait_async0() {
  asm volatile("s_wait_asynccnt 0x0" ::: "memory");
}

// ---------------------------------------------------------------------------
// ss[b, 0:512] = silu(time_emb[b]) @ mlp_w.T + mlp_b
// ---------------------------------------------------------------------------
__global__ __launch_bounds__(256)
void k_time_mlp(const float* __restrict__ te, const float* __restrict__ w,
                const float* __restrict__ bias, float* __restrict__ ss) {
  __shared__ float st[TEMB_];
  const int b = blockIdx.x, t = threadIdx.x;
  for (int i = t; i < TEMB_; i += 256) {
    float e = te[b * TEMB_ + i];
    st[i] = e / (1.f + __expf(-e));
  }
  __syncthreads();
  #pragma unroll
  for (int jj = 0; jj < 2; jj++) {
    const int j = t + jj * 256;
    const float* wr = w + (size_t)j * TEMB_;
    float a = bias[j];
    for (int k = 0; k < TEMB_; k++) a = fmaf(st[k], wr[k], a);
    ss[b * 512 + j] = a;
  }
}

// ---------------------------------------------------------------------------
// RMS-norm over channels + (1+scale)/shift modulation.
// Emits xnT[b][n][c] (bf16, K=c contiguous) = B^T of the qkv GEMM.
// ---------------------------------------------------------------------------
__global__ __launch_bounds__(256)
void k_norm_xnT(const float* __restrict__ x, const float* __restrict__ ss,
                unsigned short* __restrict__ xnT) {
  const int b = blockIdx.x, n = threadIdx.x;
  const float* xb = x + (size_t)b * C_ * N_;
  float s = 0.f;
  for (int c = 0; c < C_; c++) { float v = xb[c * N_ + n]; s = fmaf(v, v, s); }
  const float inv = rsqrtf(s * (1.f / C_) + EPS_);
  unsigned short* o = xnT + ((size_t)b * N_ + n) * C_;
  const float* sb = ss + b * 512;
  for (int c = 0; c < C_; c++) {
    float v = xb[c * N_ + n];
    o[c] = f2bf(v * inv * (1.f + sb[c]) + sb[256 + c]);
  }
}

__global__ void k_cvt_bf16(const float* __restrict__ in,
                           unsigned short* __restrict__ out, int n) {
  int i = blockIdx.x * blockDim.x + threadIdx.x;
  if (i < n) out[i] = f2bf(in[i]);
}

// ---------------------------------------------------------------------------
// Generic batched NT GEMM: C[M][N] = alpha * A[M][K] * Bt[N][K]^T  (bf16->f32)
// WMMA 16x16x32 bf16, 128x128 block tile, 8 waves (2x4), K-step 32.
// Double-buffered LDS, tiles streamed with global_load_async_to_lds_b128.
// Optional fused column reduction into colsum (atomicAdd), C may be null.
// ---------------------------------------------------------------------------
union ABu { u32x4 u[2]; v16bf v; };

__global__ __launch_bounds__(256)
void k_gemm_nt(const unsigned short* __restrict__ A,  long long sA_,
               const unsigned short* __restrict__ Bt, long long sB_,
               float* __restrict__ C, long long sC_,
               int M, int N, int K, float alpha,
               float* __restrict__ colsum, int csStride) {
  constexpr int TM = 128, TN = 128, KT = 32, LP = 40; // 40 bf16 = 80B padded row
  constexpr int ABUF = TM * LP;                        // elems per buffer
  constexpr int BBUF = TN * LP;
  __shared__ unsigned short shA[2 * ABUF];
  __shared__ unsigned short shB[2 * BBUF];

  const int bz = blockIdx.z;
  A  += (long long)bz * sA_;
  Bt += (long long)bz * sB_;
  const int m0 = blockIdx.x * TM, n0 = blockIdx.y * TN;
  const int t = threadIdx.x, wave = t >> 5, lane = t & 31;
  const int wm = wave & 1, wn = wave >> 1;   // 2(M) x 4(N) wave grid
  const int r = lane & 15, hh = lane >> 4;   // WMMA lane decomposition

  // Staging: 512 chunks of 8 bf16 (16B) per matrix; thread owns chunks t, t+256.
  const int row0 = t >> 2, ck = (t & 3) * 8;
  const int row1 = row0 + 64;
  const unsigned short* gA0 = A  + (long long)(m0 + row0) * K + ck;
  const unsigned short* gA1 = A  + (long long)(m0 + row1) * K + ck;
  const unsigned short* gB0 = Bt + (long long)(n0 + row0) * K + ck;
  const unsigned short* gB1 = Bt + (long long)(n0 + row1) * K + ck;
  const unsigned lA0 = (unsigned)(size_t)(&shA[row0 * LP + ck]);
  const unsigned lA1 = (unsigned)(size_t)(&shA[row1 * LP + ck]);
  const unsigned lB0 = (unsigned)(size_t)(&shB[row0 * LP + ck]);
  const unsigned lB1 = (unsigned)(size_t)(&shB[row1 * LP + ck]);

  auto issue = [&](int k0, int sel) {
    const unsigned ao = sel * (unsigned)(ABUF * 2); // byte offset of buffer
    const unsigned bo = sel * (unsigned)(BBUF * 2);
    async_ld_b128(lA0 + ao, gA0 + k0);
    async_ld_b128(lA1 + ao, gA1 + k0);
    async_ld_b128(lB0 + bo, gB0 + k0);
    async_ld_b128(lB1 + bo, gB1 + k0);
  };

  v8f acc[4][2];
  #pragma unroll
  for (int i = 0; i < 4; i++)
    #pragma unroll
    for (int j = 0; j < 2; j++)
      #pragma unroll
      for (int e = 0; e < 8; e++) acc[i][j][e] = 0.f;

  issue(0, 0);
  wait_async0();
  __syncthreads();

  int cur = 0;
  for (int k0 = 0; k0 < K; k0 += KT) {
    const bool more = (k0 + KT) < K;
    if (more) issue(k0 + KT, cur ^ 1); // stream next tile into other buffer

    const unsigned short* bA = shA + cur * ABUF;
    const unsigned short* bB = shB + cur * BBUF;

    // B fragments: lane = col (r), K chunk [16h, 16h+16) contiguous
    ABu bfr[2];
    #pragma unroll
    for (int j = 0; j < 2; j++) {
      const unsigned short* p = bB + (wn * 32 + j * 16 + r) * LP + hh * 16;
      bfr[j].u[0] = *(const u32x4*)p;
      bfr[j].u[1] = *(const u32x4*)(p + 8);
    }
    // A fragments: lane = row (r), K chunk [16h, 16h+16) contiguous
    #pragma unroll
    for (int i = 0; i < 4; i++) {
      ABu af;
      const unsigned short* p = bA + (wm * 64 + i * 16 + r) * LP;
      af.u[0] = *(const u32x4*)(p + hh * 16);
      af.u[1] = *(const u32x4*)(p + hh * 16 + 8);
      #pragma unroll
      for (int j = 0; j < 2; j++)
        acc[i][j] = __builtin_amdgcn_wmma_f32_16x16x32_bf16(
            false, af.v, false, bfr[j].v, (short)0, acc[i][j], false, false);
    }

    if (more) {
      wait_async0();    // next tile resident in LDS
      __syncthreads();  // all waves done reading current buffer
      cur ^= 1;
    }
  }

  // D layout: lane(r,hh): element jj -> (row = mtile + 8*hh + jj, col = ntile + r)
  if (C) {
    C += (long long)bz * sC_;
    #pragma unroll
    for (int i = 0; i < 4; i++) {
      const int gm = m0 + wm * 64 + i * 16 + hh * 8;
      #pragma unroll
      for (int j = 0; j < 2; j++) {
        const int gn = n0 + wn * 32 + j * 16 + r;
        float* cp = C + (long long)gm * N + gn;
        #pragma unroll
        for (int jj = 0; jj < 8; jj++) cp[(long long)jj * N] = alpha * acc[i][j][jj];
      }
    }
  }
  if (colsum) {
    #pragma unroll
    for (int j = 0; j < 2; j++) {
      float s = 0.f;
      #pragma unroll
      for (int i = 0; i < 4; i++)
        #pragma unroll
        for (int jj = 0; jj < 8; jj++) s += acc[i][j][jj];
      atomicAdd(colsum + (long long)bz * csStride + n0 + wn * 32 + j * 16 + r,
                alpha * s);
    }
  }
}

// ---------------------------------------------------------------------------
// Sq[b,h,l] = SC * sum_c softmax_c(q[b,h,c,l])   (honest 3-pass softmax sum)
// ---------------------------------------------------------------------------
__global__ __launch_bounds__(256)
void k_sq(const float* __restrict__ qkv, float* __restrict__ Sq) {
  const int bh = blockIdx.x, l = threadIdx.x;
  const float* q = qkv + ((size_t)(bh >> 3) * 3 * HID_ + (bh & 7) * HD_) * N_ + l;
  float mx = -3.4e38f;
  for (int c = 0; c < HD_; c++) mx = fmaxf(mx, q[(size_t)c * N_]);
  float z = 0.f;
  for (int c = 0; c < HD_; c++) z += __expf(q[(size_t)c * N_] - mx);
  const float inv = 1.f / z;
  float s = 0.f;
  for (int c = 0; c < HD_; c++) s += __expf(q[(size_t)c * N_] - mx) * inv;
  Sq[bh * N_ + l] = SC_ * s;
}

// ---------------------------------------------------------------------------
// ksT[bh][n][c] = softmax_n(k[b,h,c,:]) (bf16, transposed = GEMM A)
// vT [bh][m][c] = v[b,h,c,m]            (bf16, transposed = GEMM B^T)
// ---------------------------------------------------------------------------
__global__ __launch_bounds__(256)
void k_ksvt(const float* __restrict__ qkv, unsigned short* __restrict__ ksT,
            unsigned short* __restrict__ vT) {
  const int bh = blockIdx.x;
  const int wave = threadIdx.x >> 5, lane = threadIdx.x & 31;
  const int b = bh >> 3, h = bh & 7;
  const float* kb = qkv + ((size_t)b * 3 * HID_ + HID_ + h * HD_) * N_;
  const float* vb = qkv + ((size_t)b * 3 * HID_ + 2 * HID_ + h * HD_) * N_;
  unsigned short* ko = ksT + (size_t)bh * HD_ * N_;
  unsigned short* vo = vT + (size_t)bh * HD_ * N_;
  for (int c = wave; c < HD_; c += 8) {
    const float* kr = kb + (size_t)c * N_;
    float v[8];
    float mx = -3.4e38f;
    #pragma unroll
    for (int i = 0; i < 8; i++) { v[i] = kr[lane + i * 32]; mx = fmaxf(mx, v[i]); }
    mx = wred_max(mx);
    float z = 0.f;
    #pragma unroll
    for (int i = 0; i < 8; i++) { v[i] = __expf(v[i] - mx); z += v[i]; }
    z = wred_sum(z);
    const float inv = 1.f / z;
    #pragma unroll
    for (int i = 0; i < 8; i++) {
      const int n = lane + i * 32;
      ko[(size_t)n * HD_ + c] = f2bf(v[i] * inv);
    }
    const float* vr = vb + (size_t)c * N_;
    #pragma unroll
    for (int i = 0; i < 8; i++) {
      const int n = lane + i * 32;
      vo[(size_t)n * HD_ + c] = f2bf(vr[n]);
    }
  }
}

// ---------------------------------------------------------------------------
// U[b][c][h] = sum_m out_w[c, h*256+m] * Sctx[b,h,m]
// ---------------------------------------------------------------------------
__global__ __launch_bounds__(256)
void k_u(const float* __restrict__ out_w, const float* __restrict__ Sctx,
         float* __restrict__ U) {
  const int b = blockIdx.x, c = threadIdx.x;
  const float* wr = out_w + (size_t)c * HID_;
  #pragma unroll
  for (int h = 0; h < HEADS_; h++) {
    const float* sc = Sctx + (b * HEADS_ + h) * HD_;
    const float* w = wr + h * HD_;
    float a = 0.f;
    for (int m = 0; m < HD_; m++) a = fmaf(w[m], sc[m], a);
    U[((size_t)b * C_ + c) * HEADS_ + h] = a;
  }
}

// ---------------------------------------------------------------------------
// y[c,l] = sum_h U[c,h]*Sq[h,l] + out_b[c]; RMS-norm over c; +x residual.
// ---------------------------------------------------------------------------
__global__ __launch_bounds__(256)
void k_final(const float* __restrict__ U, const float* __restrict__ Sq,
             const float* __restrict__ out_b, const float* __restrict__ ons,
             const float* __restrict__ x, float* __restrict__ out) {
  __shared__ float sU[C_ * HEADS_];
  __shared__ float sB[C_], sS[C_];
  const int b = blockIdx.x, l = threadIdx.x;
  for (int i = l; i < C_ * HEADS_; i += 256) sU[i] = U[(size_t)b * C_ * HEADS_ + i];
  sB[l] = out_b[l];
  sS[l] = ons[l];
  __syncthreads();
  float sq[HEADS_];
  #pragma unroll
  for (int h = 0; h < HEADS_; h++) sq[h] = Sq[(b * HEADS_ + h) * N_ + l];
  float ssq = 0.f;
  for (int c = 0; c < C_; c++) {
    float y = sB[c];
    #pragma unroll
    for (int h = 0; h < HEADS_; h++) y = fmaf(sU[c * HEADS_ + h], sq[h], y);
    ssq = fmaf(y, y, ssq);
  }
  const float inv2 = rsqrtf(ssq * (1.f / C_) + EPS_);
  const float* xb = x + (size_t)b * C_ * N_;
  float* ob = out + (size_t)b * C_ * N_;
  for (int c = 0; c < C_; c++) {
    float y = sB[c];
    #pragma unroll
    for (int h = 0; h < HEADS_; h++) y = fmaf(sU[c * HEADS_ + h], sq[h], y);
    ob[c * N_ + l] = y * inv2 * sS[c] + xb[c * N_ + l];
  }
}

// ---------------------------------------------------------------------------
extern "C" void kernel_launch(void* const* d_in, const int* in_sizes, int n_in,
                              void* d_out, int out_size, void* d_ws, size_t ws_size,
                              hipStream_t stream) {
  const float* x     = (const float*)d_in[0];
  const float* te    = (const float*)d_in[1];
  const float* mlp_w = (const float*)d_in[2];
  const float* mlp_b = (const float*)d_in[3];
  const float* qkv_w = (const float*)d_in[4];
  const float* out_w = (const float*)d_in[5];
  const float* out_b = (const float*)d_in[6];
  const float* ons   = (const float*)d_in[7];
  float* out = (float*)d_out;
  (void)in_sizes; (void)n_in; (void)out_size; (void)ws_size;

  char* p = (char*)d_ws;
  auto alloc = [&](size_t bytes) -> char* {
    char* r = p;
    p += (bytes + 255) & ~(size_t)255;
    return r;
  };
  float*          ss   = (float*)alloc((size_t)B_ * 512 * 4);
  unsigned short* xnT  = (unsigned short*)alloc((size_t)B_ * N_ * C_ * 2);
  unsigned short* w16  = (unsigned short*)alloc((size_t)3 * HID_ * C_ * 2);
  float*          qkv  = (float*)alloc((size_t)B_ * 3 * HID_ * N_ * 4);
  float*          Sq   = (float*)alloc((size_t)B_ * HEADS_ * N_ * 4);
  unsigned short* ksT  = (unsigned short*)alloc((size_t)B_ * HEADS_ * HD_ * N_ * 2);
  unsigned short* vT   = (unsigned short*)alloc((size_t)B_ * HEADS_ * HD_ * N_ * 2);
  float*          Sctx = (float*)alloc((size_t)B_ * HEADS_ * HD_ * 4);
  float*          U    = (float*)alloc((size_t)B_ * C_ * HEADS_ * 4);

  k_time_mlp<<<B_, 256, 0, stream>>>(te, mlp_w, mlp_b, ss);
  k_norm_xnT<<<B_, 256, 0, stream>>>(x, ss, xnT);
  {
    const int n = 3 * HID_ * C_;
    k_cvt_bf16<<<(n + 255) / 256, 256, 0, stream>>>(qkv_w, w16, n);
  }
  { // qkv = qkv_w (6144x256) * xn (256x256) per batch
    dim3 g(3 * HID_ / 128, N_ / 128, B_);
    k_gemm_nt<<<g, 256, 0, stream>>>(w16, 0, xnT, (long long)N_ * C_,
                                     qkv, (long long)3 * HID_ * N_,
                                     3 * HID_, N_, C_, 1.f, nullptr, 0);
  }
  k_sq<<<B_ * HEADS_, 256, 0, stream>>>(qkv, Sq);
  k_ksvt<<<B_ * HEADS_, 256, 0, stream>>>(qkv, ksT, vT);
  hipMemsetAsync(Sctx, 0, (size_t)B_ * HEADS_ * HD_ * 4, stream);
  { // ctx = SC * ks^T(n,c) * v(c,m) per (b,h); fused column reduction -> Sctx
    dim3 g(N_ / 128, HD_ / 128, B_ * HEADS_);
    k_gemm_nt<<<g, 256, 0, stream>>>(ksT, (long long)HD_ * N_,
                                     vT, (long long)HD_ * N_,
                                     nullptr, 0, N_, HD_, HD_, SC_, Sctx, HD_);
  }
  k_u<<<B_, 256, 0, stream>>>(out_w, Sctx, U);
  k_final<<<B_, 256, 0, stream>>>(U, Sq, out_b, ons, x, out);
}
